// FullyAwareAttention_64209761075496
// MI455X (gfx1250) — matrix operand order repaired and checked by reading, not compiled, not run on hardware
//
#include <hip/hip_runtime.h>
#include <hip/hip_bf16.h>
#include <math.h>

// ---------------------------------------------------------------------------
// FullyAwareAttention on MI455X (gfx1250, wave32, WMMA f16->f32)
// B=16, L1=L2=1024, D=1024
// ---------------------------------------------------------------------------

typedef __attribute__((ext_vector_type(16))) _Float16 v16h;
typedef __attribute__((ext_vector_type(8)))  _Float16 v8h;
typedef __attribute__((ext_vector_type(8)))  float    v8f;

#define FAA_B  16
#define FAA_L  1024   // L1 == L2
#define FAA_D  1024

#define TM 128
#define TN 128
#define TK 32
#define LDT 40        // TK + 8 halves pad -> 80B row stride, conflict-free

// ---------------------------------------------------------------------------
// Fragment loads from LDS tiles (layouts per CDNA5 ISA 7.12.2)
// A 16x32 f16: lanes 0-15 -> M=lane, K {0..7,16..23}; lanes 16-31 -> K {8..15,24..31}
// B 32x16 f16: lane -> N=lane&15; same K order as A (lanes 0-15 K {0..7,16..23}; lanes 16-31 K {8..15,24..31})
// ---------------------------------------------------------------------------
__device__ __forceinline__ v16h frag_a(const _Float16* As, int row, int laneHi) {
    const _Float16* p = As + row * LDT + laneHi * 8;
    v8h lo = *(const v8h*)(p);
    v8h hi = *(const v8h*)(p + 16);
    v16h a;
#pragma unroll
    for (int i = 0; i < 8; ++i) { a[i] = lo[i]; a[i + 8] = hi[i]; }
    return a;
}

__device__ __forceinline__ v16h frag_b(const _Float16* Bs, int col, int laneHi) {
    const _Float16* p = Bs + col * LDT + laneHi * 8;
    v8h lo = *(const v8h*)(p);
    v8h hi = *(const v8h*)(p + 16);
    v16h b;
#pragma unroll
    for (int i = 0; i < 8; ++i) { b[i] = lo[i]; b[i + 8] = hi[i]; }
    return b;
}

// One K-step for a wave's 32x64 tile: 2 A-frags x 4 B-frags -> 8 WMMAs
__device__ __forceinline__ void wave_mma(const _Float16* As, const _Float16* Bs,
                                         v8f acc[2][4], int wm, int wn,
                                         int lane16, int laneHi) {
    v16h a[2], b[4];
#pragma unroll
    for (int mf = 0; mf < 2; ++mf)
        a[mf] = frag_a(As, wm * 32 + mf * 16 + lane16, laneHi);
#pragma unroll
    for (int nf = 0; nf < 4; ++nf)
        b[nf] = frag_b(Bs, wn * 64 + nf * 16 + lane16, laneHi);
#pragma unroll
    for (int mf = 0; mf < 2; ++mf)
#pragma unroll
        for (int nf = 0; nf < 4; ++nf)
            acc[mf][nf] = __builtin_amdgcn_wmma_f32_16x16x32_f16(
                false, a[mf], false, b[nf], (short)0, acc[mf][nf], false, false);
}

// Float atomic max via integer bit tricks (monotone encodings; init must be <0)
__device__ __forceinline__ void atomicMaxF32(float* addr, float val) {
    if (val >= 0.f) atomicMax((int*)addr, __float_as_int(val));
    else            atomicMin((unsigned int*)addr, __float_as_uint(val));
}

// ---------------------------------------------------------------------------
// K0: init column stats
// ---------------------------------------------------------------------------
__global__ void k_init_stats(float* __restrict__ colmax, float* __restrict__ colsum) {
    int i = blockIdx.x * 256 + threadIdx.x;
    colmax[i] = -3.402823466e38f;
    colsum[i] = 0.f;
}

// ---------------------------------------------------------------------------
// K1: Wt[e][d] = (f16) W[d][e]   (so GEMM1 B-fragments read contiguous K)
// ---------------------------------------------------------------------------
__global__ void k_transpose_w(const float* __restrict__ W, _Float16* __restrict__ Wt) {
    __shared__ float tile[32][33];
    int bx = blockIdx.x * 32, by = blockIdx.y * 32;
    int tx = threadIdx.x, ty = threadIdx.y;            // block (32, 8)
#pragma unroll
    for (int r = 0; r < 32; r += 8)
        tile[ty + r][tx] = W[(size_t)(by + ty + r) * FAA_D + bx + tx];
    __syncthreads();
#pragma unroll
    for (int r = 0; r < 32; r += 8)
        Wt[(size_t)(bx + ty + r) * FAA_D + by + tx] = (_Float16)tile[tx][ty + r];
}

// ---------------------------------------------------------------------------
// K2: sxh = relu(x . W), syh = relu(y . W)   (f32 in, f16 out; z: 0..2B-1)
// ---------------------------------------------------------------------------
__global__ __launch_bounds__(256)
void k_gemm_xw(const float* __restrict__ x, const float* __restrict__ y,
               const _Float16* __restrict__ Wt,
               _Float16* __restrict__ sxh, _Float16* __restrict__ syh) {
    __shared__ __align__(16) _Float16 As[TM * LDT];
    __shared__ __align__(16) _Float16 Bs[TN * LDT];

    const int t = threadIdx.x;
    const int lane = t & 31, lane16 = lane & 15, laneHi = lane >> 4;
    const int waveId = t >> 5, wm = waveId & 3, wn = waveId >> 2;

    const int bz = blockIdx.z;
    const float* A = (bz < FAA_B) ? (x + (size_t)bz * FAA_L * FAA_D)
                                  : (y + (size_t)(bz - FAA_B) * FAA_L * FAA_D);
    _Float16* C = ((bz < FAA_B) ? sxh : syh) + (size_t)(bz & (FAA_B - 1)) * FAA_L * FAA_D;

    const int m0 = blockIdx.y * TM;
    const int n0 = blockIdx.x * TN;

    v8f acc[2][4] = {};

#pragma unroll 1
    for (int k0 = 0; k0 < FAA_D; k0 += TK) {
        // A tile: 128x32 f32 -> f16 (1024 float4 loads)
#pragma unroll
        for (int it = 0; it < 4; ++it) {
            int idx = t + it * 256;
            int row = idx >> 3;
            int c4  = (idx & 7) * 4;
            const float4 v = *(const float4*)(A + (size_t)(m0 + row) * FAA_D + k0 + c4);
            _Float16* d = As + row * LDT + c4;
            d[0] = (_Float16)v.x; d[1] = (_Float16)v.y;
            d[2] = (_Float16)v.z; d[3] = (_Float16)v.w;
        }
        // B tile: 128x32 f16 from Wt (N-major, contiguous K)
#pragma unroll
        for (int it = 0; it < 2; ++it) {
            int idx = t + it * 256;
            int row = idx >> 2;
            int c8  = (idx & 3) * 8;
            v8h v = *(const v8h*)(Wt + (size_t)(n0 + row) * FAA_D + k0 + c8);
            *(v8h*)(Bs + row * LDT + c8) = v;
        }
        if (k0 + TK < FAA_D)
            __builtin_prefetch(A + (size_t)(m0 + (t >> 3)) * FAA_D + k0 + TK, 0, 1);
        __syncthreads();
        wave_mma(As, Bs, acc, wm, wn, lane16, laneHi);
        __syncthreads();
    }

    // relu + f16 store
#pragma unroll
    for (int mf = 0; mf < 2; ++mf)
#pragma unroll
        for (int nf = 0; nf < 4; ++nf)
#pragma unroll
            for (int r = 0; r < 8; ++r) {
                int m = m0 + wm * 32 + mf * 16 + laneHi * 8 + r;
                int n = n0 + wn * 64 + nf * 16 + lane16;
                float v = acc[mf][nf][r];
                C[(size_t)m * FAA_D + n] = (_Float16)(v > 0.f ? v : 0.f);
            }
}

// ---------------------------------------------------------------------------
// K3: s = sxh . syh^T  (f16 in, f32 out) + fused per-column (over i) max
// ---------------------------------------------------------------------------
__global__ __launch_bounds__(256)
void k_gemm_s(const _Float16* __restrict__ sxh, const _Float16* __restrict__ syh,
              float* __restrict__ s, float* __restrict__ colmax) {
    __shared__ __align__(16) _Float16 As[TM * LDT];
    __shared__ __align__(16) _Float16 Bs[TN * LDT];

    const int t = threadIdx.x;
    const int lane = t & 31, lane16 = lane & 15, laneHi = lane >> 4;
    const int waveId = t >> 5, wm = waveId & 3, wn = waveId >> 2;

    const int b  = blockIdx.z;
    const _Float16* A  = sxh + (size_t)b * FAA_L * FAA_D;
    const _Float16* Bt = syh + (size_t)b * FAA_L * FAA_D;   // Bt[n][k] directly

    const int m0 = blockIdx.y * TM;
    const int n0 = blockIdx.x * TN;

    v8f acc[2][4] = {};

#pragma unroll 1
    for (int k0 = 0; k0 < FAA_D; k0 += TK) {
#pragma unroll
        for (int it = 0; it < 2; ++it) {
            int idx = t + it * 256;
            int row = idx >> 2;
            int c8  = (idx & 3) * 8;
            *(v8h*)(As + row * LDT + c8) =
                *(const v8h*)(A + (size_t)(m0 + row) * FAA_D + k0 + c8);
            *(v8h*)(Bs + row * LDT + c8) =
                *(const v8h*)(Bt + (size_t)(n0 + row) * FAA_D + k0 + c8);
        }
        if (k0 + TK < FAA_D)
            __builtin_prefetch(A + (size_t)(m0 + (t >> 2)) * FAA_D + k0 + TK, 0, 1);
        __syncthreads();
        wave_mma(As, Bs, acc, wm, wn, lane16, laneHi);
        __syncthreads();
    }

    float* S = s + (size_t)b * FAA_L * FAA_L;
#pragma unroll
    for (int nf = 0; nf < 4; ++nf) {
        int n = n0 + wn * 64 + nf * 16 + lane16;
        float mx = -3.402823466e38f;
#pragma unroll
        for (int mf = 0; mf < 2; ++mf)
#pragma unroll
            for (int r = 0; r < 8; ++r) {
                int m = m0 + wm * 32 + mf * 16 + laneHi * 8 + r;
                float v = acc[mf][nf][r];
                S[(size_t)m * FAA_L + n] = v;
                mx = fmaxf(mx, v);
            }
        atomicMaxF32(&colmax[b * FAA_L + n], mx);
    }
}

// ---------------------------------------------------------------------------
// K4: s <- exp(s - colmax[j]) * mask[j]; colsum[j] += partial column sums
// ---------------------------------------------------------------------------
__global__ __launch_bounds__(256)
void k_softmax(float* __restrict__ s, const float* __restrict__ mask,
               const float* __restrict__ colmax, float* __restrict__ colsum) {
    const int b  = blockIdx.z;
    const int j  = blockIdx.x * 256 + threadIdx.x;
    const int i0 = blockIdx.y * 128;
    const float cmax = colmax[b * FAA_L + j];
    const float mval = mask[b * FAA_L + j];
    float* S = s + (size_t)b * FAA_L * FAA_L;
    float sum = 0.f;
#pragma unroll 4
    for (int i = 0; i < 128; ++i) {
        size_t off = (size_t)(i0 + i) * FAA_L + j;
        float e = __expf(S[off] - cmax) * mval;
        S[off] = e;
        sum += e;
    }
    atomicAdd(&colsum[b * FAA_L + j], sum);
}

// ---------------------------------------------------------------------------
// K5: out = (s * rcp(colsum+eps)) . V   (scale folded into A conversion,
//     V transposed through LDS in the B-tile fill)
// ---------------------------------------------------------------------------
__global__ __launch_bounds__(256)
void k_gemm_out(const float* __restrict__ s, const float* __restrict__ values,
                const float* __restrict__ colsum, float* __restrict__ out) {
    __shared__ __align__(16) _Float16 As[TM * LDT];
    __shared__ __align__(16) _Float16 Bs[TN * LDT];

    const int t = threadIdx.x;
    const int lane = t & 31, lane16 = lane & 15, laneHi = lane >> 4;
    const int waveId = t >> 5, wm = waveId & 3, wn = waveId >> 2;

    const int b = blockIdx.z;
    const float* Sa = s      + (size_t)b * FAA_L * FAA_L;   // A: [L1][L2]
    const float* Vv = values + (size_t)b * FAA_L * FAA_D;   // B: [L2][D]
    const float* cs = colsum + (size_t)b * FAA_L;

    const int m0 = blockIdx.y * TM;
    const int n0 = blockIdx.x * TN;

    v8f acc[2][4] = {};

#pragma unroll 1
    for (int k0 = 0; k0 < FAA_L; k0 += TK) {
        // A tile: alpha = s * rcp(colsum+eps), f32 -> f16
#pragma unroll
        for (int it = 0; it < 4; ++it) {
            int idx = t + it * 256;
            int row = idx >> 3;
            int c4  = (idx & 7) * 4;
            const float4 v = *(const float4*)(Sa + (size_t)(m0 + row) * FAA_L + k0 + c4);
            float r0 = __builtin_amdgcn_rcpf(cs[k0 + c4 + 0] + 1e-10f);
            float r1 = __builtin_amdgcn_rcpf(cs[k0 + c4 + 1] + 1e-10f);
            float r2 = __builtin_amdgcn_rcpf(cs[k0 + c4 + 2] + 1e-10f);
            float r3 = __builtin_amdgcn_rcpf(cs[k0 + c4 + 3] + 1e-10f);
            _Float16* d = As + row * LDT + c4;
            d[0] = (_Float16)(v.x * r0); d[1] = (_Float16)(v.y * r1);
            d[2] = (_Float16)(v.z * r2); d[3] = (_Float16)(v.w * r3);
        }
        // B tile (transposed): Bs[n][k] = V[k0+k][n0+n]
#pragma unroll
        for (int it = 0; it < 4; ++it) {
            int idx = t + it * 256;
            int kk = idx >> 5;          // 0..31
            int c4 = (idx & 31) * 4;    // 0..124
            const float4 v = *(const float4*)(Vv + (size_t)(k0 + kk) * FAA_D + n0 + c4);
            Bs[(c4 + 0) * LDT + kk] = (_Float16)v.x;
            Bs[(c4 + 1) * LDT + kk] = (_Float16)v.y;
            Bs[(c4 + 2) * LDT + kk] = (_Float16)v.z;
            Bs[(c4 + 3) * LDT + kk] = (_Float16)v.w;
        }
        if (k0 + TK < FAA_L)
            __builtin_prefetch(Sa + (size_t)(m0 + (t >> 3)) * FAA_L + k0 + TK, 0, 1);
        __syncthreads();
        wave_mma(As, Bs, acc, wm, wn, lane16, laneHi);
        __syncthreads();
    }

    float* O = out + (size_t)b * FAA_L * FAA_D;
#pragma unroll
    for (int mf = 0; mf < 2; ++mf)
#pragma unroll
        for (int nf = 0; nf < 4; ++nf)
#pragma unroll
            for (int r = 0; r < 8; ++r) {
                int m = m0 + wm * 32 + mf * 16 + laneHi * 8 + r;
                int n = n0 + wn * 64 + nf * 16 + lane16;
                O[(size_t)m * FAA_D + n] = acc[mf][nf][r];
            }
}

// ---------------------------------------------------------------------------
extern "C" void kernel_launch(void* const* d_in, const int* in_sizes, int n_in,
                              void* d_out, int out_size, void* d_ws, size_t ws_size,
                              hipStream_t stream) {
    const float* x      = (const float*)d_in[0];
    const float* y      = (const float*)d_in[1];
    const float* values = (const float*)d_in[2];
    const float* mask   = (const float*)d_in[3];
    const float* W      = (const float*)d_in[4];
    float* out = (float*)d_out;

    char* ws = (char*)d_ws;
    size_t off = 0;
    _Float16* Wt  = (_Float16*)(ws + off); off += (size_t)FAA_D * FAA_D * 2;          //  2 MiB
    _Float16* sxh = (_Float16*)(ws + off); off += (size_t)FAA_B * FAA_L * FAA_D * 2;  // 32 MiB
    _Float16* syh = (_Float16*)(ws + off); off += (size_t)FAA_B * FAA_L * FAA_D * 2;  // 32 MiB
    float*    s   = (float*)(ws + off);    off += (size_t)FAA_B * FAA_L * FAA_L * 4;  // 64 MiB
    float* colmax = (float*)(ws + off);    off += (size_t)FAA_B * FAA_L * 4;
    float* colsum = (float*)(ws + off);    off += (size_t)FAA_B * FAA_L * 4;

    k_init_stats<<<FAA_B * FAA_L / 256, 256, 0, stream>>>(colmax, colsum);
    k_transpose_w<<<dim3(FAA_D / 32, FAA_D / 32), dim3(32, 8), 0, stream>>>(W, Wt);
    k_gemm_xw<<<dim3(FAA_L / TN, FAA_L / TM, 2 * FAA_B), 256, 0, stream>>>(x, y, Wt, sxh, syh);
    k_gemm_s<<<dim3(FAA_L / TN, FAA_L / TM, FAA_B), 256, 0, stream>>>(sxh, syh, s, colmax);
    k_softmax<<<dim3(FAA_L / 256, FAA_L / 128, FAA_B), 256, 0, stream>>>(s, mask, colmax, colsum);
    k_gemm_out<<<dim3(FAA_D / TN, FAA_L / TM, FAA_B), 256, 0, stream>>>(s, values, colsum, out);
}
